// Seq2SeqModel_47150150975929
// MI455X (gfx1250) — matrix operand order repaired and checked by reading, not compiled, not run on hardware
//
#include <hip/hip_runtime.h>
#include <hip/hip_bf16.h>
#include <cstdint>

#define BB 32
#define SS 64
#define TT 64
#define EE 256
#define HH 512
#define H2 1024
#define H4 2048
#define VTC 32000

typedef __attribute__((ext_vector_type(16))) _Float16 v16h;
typedef __attribute__((ext_vector_type(8)))  _Float16 v8h;
typedef __attribute__((ext_vector_type(8)))  float    v8f;

// ---------------- WMMA fragment loads (ISA 7.12.2 16-bit layouts, wave32) ----------------
// A (16x32, MxK): lanes 0-15 row M=lane: K = kb+0..7 then kb+16..23
//                 lanes 16-31 row M=lane-16: K = kb+8..15 then kb+24..31
__device__ __forceinline__ v16h load_a_frag(const _Float16* __restrict__ A, int lda,
                                            int row0, int kb, int lane) {
  int r  = row0 + (lane & 15);
  int ko = (lane & 16) ? 8 : 0;
  const _Float16* p = A + (size_t)r * lda + kb + ko;
  v8h lo = *(const v8h*)p;
  v8h hi = *(const v8h*)(p + 16);
  v16h o;
#pragma unroll
  for (int i = 0; i < 8; ++i) { o[i] = lo[i]; o[i + 8] = hi[i]; }
  return o;
}
// B (32x16, KxN) built from W stored row-major (N,K)  (we compute x @ W^T):
// lanes 0-15: col N=lane, K=kb+0..7 then kb+16..23 ; lanes 16-31: col N=lane-16, K=kb+8..15 then kb+24..31
__device__ __forceinline__ v16h load_b_frag(const _Float16* __restrict__ Wt, int ldw,
                                            int col0, int kb, int lane) {
  int n  = col0 + (lane & 15);
  int k2 = (lane & 16) ? 8 : 0;
  const _Float16* p = Wt + (size_t)n * ldw + kb + k2;
  v8h lo = *(const v8h*)p;
  v8h hi = *(const v8h*)(p + 16);
  v16h o;
#pragma unroll
  for (int i = 0; i < 8; ++i) { o[i] = lo[i]; o[i + 8] = hi[i]; }
  return o;
}

// ---------------- generic WMMA GEMM:  C(MxN) (+)= A(MxK,f16) * W(N,K,f16)^T + bias ------
// one wave -> one 16x32 output tile (A frag reused for two WMMAs)
__global__ __launch_bounds__(256) void k_gemm(
    const _Float16* __restrict__ A, int lda,
    const _Float16* __restrict__ W, int ldw,
    const float* __restrict__ bias,
    float* __restrict__ C, long long ldc,
    int M, int N, int K, int accum)
{
  int lane = threadIdx.x & 31;
  int wid  = threadIdx.x >> 5;
  int tile = blockIdx.x * 8 + wid;
  int nt32 = N >> 5;
  int total = (M >> 4) * nt32;
  if (tile >= total) return;
  int mt = tile / nt32, nt = tile % nt32;
  int row0 = mt << 4, col0 = nt << 5;

  v8f acc0 = {};
  v8f acc1 = {};
  for (int kb = 0; kb < K; kb += 32) {
    v16h a  = load_a_frag(A, lda, row0, kb, lane);
    v16h b0 = load_b_frag(W, ldw, col0,      kb, lane);
    v16h b1 = load_b_frag(W, ldw, col0 + 16, kb, lane);
    if (kb + 32 < K) { // pull next K-chunk of the streamed weight into cache
      __builtin_prefetch(W + (size_t)(col0 + (lane & 15)) * ldw + kb + 32, 0, 1);
      __builtin_prefetch(W + (size_t)(col0 + 16 + (lane & 15)) * ldw + kb + 32, 0, 1);
    }
    acc0 = __builtin_amdgcn_wmma_f32_16x16x32_f16(false, a, false, b0, (short)0, acc0, false, false);
    acc1 = __builtin_amdgcn_wmma_f32_16x16x32_f16(false, a, false, b1, (short)0, acc1, false, false);
  }
  // C/D layout: lanes 0-15 col n, rows v ; lanes 16-31 col n, rows v+8
  int n  = lane & 15;
  int rb = (lane & 16) ? 8 : 0;
  float bz0 = bias ? bias[col0 + n]      : 0.f;
  float bz1 = bias ? bias[col0 + 16 + n] : 0.f;
#pragma unroll
  for (int v = 0; v < 8; ++v) {
    size_t r  = (size_t)(row0 + rb + v);
    size_t i0 = r * (size_t)ldc + col0 + n;
    size_t i1 = i0 + 16;
    float v0 = acc0[v] + bz0;
    float v1 = acc1[v] + bz1;
    if (accum) { v0 += C[i0]; v1 += C[i1]; }
    C[i0] = v0; C[i1] = v1;
  }
}

// ---------------- elementwise / utility kernels ----------------
__global__ __launch_bounds__(256) void k_cast(const float* __restrict__ s,
                                              _Float16* __restrict__ d, long long n) {
  for (long long i = (long long)blockIdx.x * 256 + threadIdx.x; i < n;
       i += (long long)gridDim.x * 256)
    d[i] = (_Float16)s[i];
}
__global__ __launch_bounds__(256) void k_zero32(float* p, long long n) {
  for (long long i = (long long)blockIdx.x * 256 + threadIdx.x; i < n;
       i += (long long)gridDim.x * 256) p[i] = 0.f;
}
__global__ __launch_bounds__(256) void k_zero16(_Float16* p, long long n) {
  for (long long i = (long long)blockIdx.x * 256 + threadIdx.x; i < n;
       i += (long long)gridDim.x * 256) p[i] = (_Float16)0.f;
}
__global__ __launch_bounds__(256) void k_addbias(const float* a, const float* b,
                                                 float* o, int n) {
  int i = blockIdx.x * 256 + threadIdx.x;
  if (i < n) o[i] = a[i] + b[i];
}
// gather embeddings, time-major out[t][b][e] (f16)
__global__ __launch_bounds__(256) void k_embed(const int* __restrict__ tok, int tokLd,
                                               const float* __restrict__ emb,
                                               _Float16* __restrict__ out, long long total) {
  for (long long idx = (long long)blockIdx.x * 256 + threadIdx.x; idx < total;
       idx += (long long)gridDim.x * 256) {
    int e = (int)(idx % EE);
    long long r = idx / EE;
    int b = (int)(r % BB);
    int t = (int)(r / BB);
    int token = tok[b * tokLd + t];
    out[idx] = (_Float16)emb[(size_t)token * EE + e];
  }
}
__device__ __forceinline__ float sigm(float x) { return 1.f / (1.f + expf(-x)); }
// fused LSTM cell: gates G(32x2048) -> h,c update (+length masking) + f16 side outputs
__global__ __launch_bounds__(256) void k_cell(
    const float* __restrict__ G, float* __restrict__ h, float* __restrict__ c,
    const int* __restrict__ lens, int t,
    _Float16* __restrict__ h16, int ldh16,
    _Float16* __restrict__ c16, int ldc16,
    _Float16* __restrict__ y16, int ldy16,
    float* __restrict__ y32, int ldy32)
{
  int idx = blockIdx.x * 256 + threadIdx.x;
  if (idx >= BB * HH) return;
  int b = idx >> 9, d = idx & (HH - 1);
  const float* g = G + (size_t)b * H4;
  float gi = g[d], gf = g[HH + d], gg = g[2 * HH + d], go = g[3 * HH + d];
  float cn = sigm(gf) * c[idx] + sigm(gi) * tanhf(gg);
  float hn = sigm(go) * tanhf(cn);
  bool v = lens ? (t < lens[b]) : true;
  float hw = v ? hn : h[idx];
  float cw = v ? cn : c[idx];
  h[idx] = hw; c[idx] = cw;
  h16[(size_t)b * ldh16 + d] = (_Float16)hw;
  if (c16) c16[(size_t)b * ldc16 + d] = (_Float16)cw;
  if (y16) y16[(size_t)b * ldy16 + d] = (_Float16)(v ? hn : 0.f);
  if (y32) y32[(size_t)b * ldy32 + d] = v ? hn : 0.f;
}
// Bahdanau attention for one decode step: scores+softmax+context, writes xin = [e_t | ctx] (f16)
__global__ __launch_bounds__(256) void k_attn(
    const float* __restrict__ decq, const float* __restrict__ encProj,
    const float* __restrict__ encOut, const float* __restrict__ Wv,
    const float* __restrict__ bv, const int* __restrict__ lens,
    const _Float16* __restrict__ embq, int t, _Float16* __restrict__ xin)
{
  __shared__ float sc[SS];
  int b = blockIdx.x;
  int tid = threadIdx.x;
  int lane = tid & 31, wid = tid >> 5;
  const float* dq = decq + (size_t)b * H2;
#pragma unroll 1
  for (int i = 0; i < 8; ++i) {
    int tp = wid * 8 + i;
    const float* ep = encProj + ((size_t)tp * BB + b) * H2;
    float p = 0.f;
    for (int d = lane; d < H2; d += 32)
      p += tanhf(dq[d] + ep[d]) * Wv[d];
#pragma unroll
    for (int o = 16; o > 0; o >>= 1) p += __shfl_xor(p, o, 32);
    if (lane == 0) {
      float s = p + bv[0];
      sc[tp] = (tp < lens[b]) ? s : -1.0e10f;
    }
  }
  __syncthreads();
  float mx = -3.0e38f;
  for (int j = 0; j < SS; ++j) mx = fmaxf(mx, sc[j]);
  float sum = 0.f;
  for (int j = 0; j < SS; ++j) sum += expf(sc[j] - mx);
  float inv = 1.f / sum;
  for (int d = tid; d < H2; d += 256) {
    float acc = 0.f;
    for (int tp = 0; tp < SS; ++tp)
      acc += expf(sc[tp] - mx) * encOut[((size_t)tp * BB + b) * H2 + d];
    xin[(size_t)b * 1280 + EE + d] = (_Float16)(acc * inv);
  }
  xin[(size_t)b * 1280 + tid] = embq[((size_t)t * BB + b) * EE + tid]; // tid < 256 == EE
}
__global__ __launch_bounds__(256) void k_replicate(const float* __restrict__ h0,
                                                   const float* __restrict__ c0,
                                                   float* h, float* c, _Float16* h16) {
  int idx = blockIdx.x * 256 + threadIdx.x;
  if (idx >= BB * HH) return;
  float hv = h0[idx], cv = c0[idx];
#pragma unroll
  for (int l = 0; l < 4; ++l) {
    h[l * BB * HH + idx] = hv;
    c[l * BB * HH + idx] = cv;
    h16[l * BB * HH + idx] = (_Float16)hv;
  }
}
__global__ __launch_bounds__(256) void k_zero_out0(float* out) {
  int idx = blockIdx.x * 256 + threadIdx.x;
  if (idx >= BB * VTC) return;
  int b = idx / VTC, v = idx % VTC;
  out[(size_t)b * TT * VTC + v] = 0.f;
}

// ---------------- host orchestration ----------------
extern "C" void kernel_launch(void* const* d_in, const int* in_sizes, int n_in,
                              void* d_out, int out_size, void* d_ws, size_t ws_size,
                              hipStream_t stream)
{
  (void)in_sizes; (void)n_in; (void)out_size; (void)ws_size;
  const int*   src  = (const int*)d_in[0];
  const int*   tgt  = (const int*)d_in[1];
  const int*   lens = (const int*)d_in[2];
  const float* srcE = (const float*)d_in[3];
  const float* tgtE = (const float*)d_in[4];
  float* out = (float*)d_out;

  char* ws = (char*)d_ws; size_t off = 0;
  auto alloc  = [&](size_t bytes) -> char* {
    size_t o = (off + 255) & ~(size_t)255; off = o + bytes; return ws + o;
  };
  auto allocH = [&](size_t n) -> _Float16* { return (_Float16*)alloc(n * 2); };
  auto allocF = [&](size_t n) -> float*    { return (float*)alloc(n * 4); };

  auto cast = [&](const void* s, _Float16* d, size_t n) {
    long long nn = (long long)n;
    int blk = (int)((nn + 255) / 256); if (blk > 4096) blk = 4096;
    k_cast<<<blk, 256, 0, stream>>>((const float*)s, d, nn);
  };
  auto zero16 = [&](_Float16* p, size_t n) {
    long long nn = (long long)n;
    int blk = (int)((nn + 255) / 256); if (blk > 4096) blk = 4096;
    k_zero16<<<blk, 256, 0, stream>>>(p, nn);
  };
  auto zero32 = [&](float* p, size_t n) {
    long long nn = (long long)n;
    int blk = (int)((nn + 255) / 256); if (blk > 4096) blk = 4096;
    k_zero32<<<blk, 256, 0, stream>>>(p, nn);
  };
  auto gemm = [&](const _Float16* A, int lda, const _Float16* W, int ldw,
                  const float* bias, float* C, long long ldc,
                  int M, int N, int K, int accum) {
    int tiles = (M / 16) * (N / 32);
    int blocks = (tiles + 7) / 8;
    k_gemm<<<blocks, 256, 0, stream>>>(A, lda, W, ldw, bias, C, ldc, M, N, K, accum);
  };

  // ---- f16 weight conversions + combined biases ----
  int encIn[4] = {EE, EE, H2, H2};
  _Float16 *wEncWih[4], *wEncWhh[4]; float* encBias[4];
  for (int l = 0; l < 4; ++l) {
    wEncWih[l] = allocH((size_t)H4 * encIn[l]);
    wEncWhh[l] = allocH((size_t)H4 * HH);
    encBias[l] = allocF(H4);
    cast(d_in[5 + 4 * l], wEncWih[l], (size_t)H4 * encIn[l]);
    cast(d_in[6 + 4 * l], wEncWhh[l], (size_t)H4 * HH);
    k_addbias<<<8, 256, 0, stream>>>((const float*)d_in[7 + 4 * l],
                                     (const float*)d_in[8 + 4 * l], encBias[l], H4);
  }
  int decIn[4] = {EE + H2, HH, HH, HH};
  _Float16 *wDecWih[4], *wDecWhh[4]; float* decBias[4];
  for (int l = 0; l < 4; ++l) {
    wDecWih[l] = allocH((size_t)H4 * decIn[l]);
    wDecWhh[l] = allocH((size_t)H4 * HH);
    decBias[l] = allocF(H4);
    cast(d_in[31 + 4 * l], wDecWih[l], (size_t)H4 * decIn[l]);
    cast(d_in[32 + 4 * l], wDecWhh[l], (size_t)H4 * HH);
    k_addbias<<<8, 256, 0, stream>>>((const float*)d_in[33 + 4 * l],
                                     (const float*)d_in[34 + 4 * l], decBias[l], H4);
  }
  _Float16* wHid  = allocH((size_t)HH * H2);  cast(d_in[21], wHid,  (size_t)HH * H2);
  _Float16* wCell = allocH((size_t)HH * H2);  cast(d_in[23], wCell, (size_t)HH * H2);
  _Float16* wWd   = allocH((size_t)H2 * HH);  cast(d_in[25], wWd,   (size_t)H2 * HH);
  _Float16* wWe   = allocH((size_t)H2 * H2);  cast(d_in[27], wWe,   (size_t)H2 * H2);
  _Float16* wWo   = allocH((size_t)VTC * HH); cast(d_in[47], wWo,   (size_t)VTC * HH);

  // ---- activation buffers (sequence buffers time-major [t][b][d]) ----
  _Float16* x16      = allocH((size_t)SS * BB * EE);
  _Float16* x116     = allocH((size_t)SS * BB * H2);
  _Float16* encOut16 = allocH((size_t)SS * BB * H2);
  float*    encOut32 = allocF((size_t)SS * BB * H2);
  float*    encProj  = allocF((size_t)SS * BB * H2);
  float*    XW       = allocF((size_t)SS * BB * H4);   // gate buffer, all timesteps
  _Float16* embq     = allocH((size_t)(TT - 1) * BB * EE);
  _Float16* fhcat    = allocH((size_t)BB * H2);
  _Float16* fccat    = allocH((size_t)BB * H2);
  _Float16* h16l1    = allocH((size_t)BB * HH);
  float*    hEnc     = allocF((size_t)BB * HH);
  float*    cEnc     = allocF((size_t)BB * HH);
  float*    h0buf    = allocF((size_t)BB * HH);
  float*    c0buf    = allocF((size_t)BB * HH);
  float*    dech     = allocF((size_t)4 * BB * HH);
  float*    decc     = allocF((size_t)4 * BB * HH);
  _Float16* dech16   = allocH((size_t)4 * BB * HH);
  _Float16* xin16    = allocH((size_t)BB * 1280);
  float*    decq     = allocF((size_t)BB * H2);
  float*    Gbuf     = allocF((size_t)BB * H4);

  // ---- embeddings (time-major, f16) ----
  k_embed<<<2048, 256, 0, stream>>>(src, SS, srcE, x16,  (long long)SS * BB * EE);
  k_embed<<<2048, 256, 0, stream>>>(tgt, TT, tgtE, embq, (long long)(TT - 1) * BB * EE);

  zero16(fhcat, (size_t)BB * H2);
  zero16(fccat, (size_t)BB * H2);

  // ---- encoder: 4 direction passes ----
  struct Dir { int layer; int rev; int l; int colofs; };
  const Dir dirs[4] = { {1, 0, 0, 0}, {1, 1, 1, HH}, {2, 0, 2, 0}, {2, 1, 3, HH} };
  for (int di = 0; di < 4; ++di) {
    const Dir d = dirs[di];
    const _Float16* Xin = (d.layer == 1) ? x16 : x116;
    int K = encIn[d.l];
    // input-side GEMM for all timesteps at once: XW = X @ Wih^T + (bih+bhh)
    gemm(Xin, K, wEncWih[d.l], K, encBias[d.l], XW, H4, SS * BB, H4, K, 0);
    zero32(hEnc, (size_t)BB * HH);
    zero32(cEnc, (size_t)BB * HH);
    _Float16* h16; int ldh;
    if (d.layer == 1) { h16 = h16l1; ldh = HH; zero16(h16l1, (size_t)BB * HH); }
    else              { h16 = fhcat + d.colofs; ldh = H2; } // region pre-zeroed
    for (int step = 0; step < SS; ++step) {
      int t = d.rev ? (SS - 1 - step) : step;
      float* Gt = XW + (size_t)t * BB * H4;
      // recurrent GEMM accumulates into precomputed gates (stream-ordered dependency)
      gemm(h16, ldh, wEncWhh[d.l], HH, nullptr, Gt, H4, BB, H4, HH, 1);
      _Float16 *y16, *c16; float* y32;
      if (d.layer == 1) {
        y16 = x116 + (size_t)t * BB * H2 + d.colofs; c16 = nullptr; y32 = nullptr;
      } else {
        y16 = encOut16 + (size_t)t * BB * H2 + d.colofs;
        y32 = encOut32 + (size_t)t * BB * H2 + d.colofs;
        c16 = fccat + d.colofs;
      }
      k_cell<<<(BB * HH + 255) / 256, 256, 0, stream>>>(
          Gt, hEnc, cEnc, lens, t, h16, ldh, c16, H2, y16, H2, y32, H2);
    }
  }

  // ---- decoder init: h0/c0 from final bi-states, broadcast to 4 layers ----
  gemm(fhcat, H2, wHid,  H2, (const float*)d_in[22], h0buf, HH, BB, HH, H2, 0);
  gemm(fccat, H2, wCell, H2, (const float*)d_in[24], c0buf, HH, BB, HH, H2, 0);
  k_replicate<<<(BB * HH + 255) / 256, 256, 0, stream>>>(h0buf, c0buf, dech, decc, dech16);
  // attention key projection (whole sequence): encProj = enc_out @ We^T + be
  gemm(encOut16, H2, wWe, H2, (const float*)d_in[28], encProj, H2, SS * BB, H2, H2, 0);

  // ---- output t=0 slice is zeros ----
  k_zero_out0<<<(BB * VTC + 255) / 256, 256, 0, stream>>>(out);

  // ---- decoder scan over 63 steps ----
  for (int t = 0; t < TT - 1; ++t) {
    // query projection from previous top-layer h
    gemm(dech16 + 3 * BB * HH, HH, wWd, HH, (const float*)d_in[26], decq, H2, BB, H2, HH, 0);
    // attention + build xin = [e_t | ctx] (f16)
    k_attn<<<BB, 256, 0, stream>>>(decq, encProj, encOut32, (const float*)d_in[29],
                                   (const float*)d_in[30], lens, embq, t, xin16);
    for (int l = 0; l < 4; ++l) {
      const _Float16* Ain = (l == 0) ? xin16 : (dech16 + (l - 1) * BB * HH);
      int K = decIn[l];
      gemm(Ain, K, wDecWih[l], K, decBias[l], Gbuf, H4, BB, H4, K, 0);
      gemm(dech16 + l * BB * HH, HH, wDecWhh[l], HH, nullptr, Gbuf, H4, BB, H4, HH, 1);
      k_cell<<<(BB * HH + 255) / 256, 256, 0, stream>>>(
          Gbuf, dech + l * BB * HH, decc + l * BB * HH, nullptr, 0,
          dech16 + l * BB * HH, HH, nullptr, 0, nullptr, 0, nullptr, 0);
    }
    // logits straight into d_out[:, t+1, :]
    gemm(dech16 + 3 * BB * HH, HH, wWo, HH, (const float*)d_in[48],
         out + (size_t)(t + 1) * VTC, (long long)TT * VTC, BB, VTC, HH, 0);
  }
}
